// GPT_38972533244656
// MI455X (gfx1250) — compile-verified, matrix operand order checked by reading, not run on hardware
//
#include <hip/hip_runtime.h>
#include <hip/hip_bf16.h>

typedef __attribute__((ext_vector_type(16))) __bf16 v16bf;
typedef __attribute__((ext_vector_type(8)))  float  v8f;

#define NB 16
#define NTT 64
#define NIT 256
#define NT 320
#define NC 768
#define NH 12
#define NDH 64
#define NL 12
#define NM (NB*NT)          // 5120 rows
#define NFF 3072
#define NV 12288

__device__ __forceinline__ unsigned short f2bf(float f) {
    unsigned u = __float_as_uint(f);
    unsigned r = u + 0x7FFFu + ((u >> 16) & 1u);
    return (unsigned short)(r >> 16);
}
__device__ __forceinline__ float bf2f(unsigned short h) {
    return __uint_as_float(((unsigned)h) << 16);
}

// ISA 16-bit A/B fragment swizzle: element k (0..31) of a 16-row subtile for
// half-group 'half' lives at lane' = (r&15)+16*((k>>3)&1), slot j below.
// Within an aligned group of 8 k-values, j is consecutive and lane' constant,
// so 16-byte global chunks map to 16-byte LDS chunks (async B128 friendly).
__device__ __forceinline__ int frag_j(int kk) { return ((kk >> 4) << 3) | (kk & 7); }
__device__ __forceinline__ int frag_kk(int half, int j) {
    return ((j >> 3) << 4) + half * 8 + (j & 7);
}

union Frag { v16bf v; unsigned short u[16]; int4 q[2]; };

// CDNA5 async global->LDS B128 copy (ASYNCcnt-tracked), ISA §15.18.3 op 98.
__device__ __forceinline__ void async_b128(void* lds, const void* gsrc) {
    unsigned l = (unsigned)(size_t)lds;  // generic->LDS offset: low 32 bits
    asm volatile("global_load_async_to_lds_b128 %0, %1, off"
                 :: "v"(l), "v"(gsrc) : "memory");
}
__device__ __forceinline__ void wait_async0() {
    asm volatile("s_wait_asynccnt 0x0" ::: "memory");
}

// ---------------------------------------------------------------- embeddings
__global__ void k_embed(const int* __restrict__ tt, const int* __restrict__ it,
                        const float* __restrict__ tte, const float* __restrict__ ite,
                        const float* __restrict__ tpe, const float* __restrict__ ipe,
                        float* __restrict__ x) {
    int idx = blockIdx.x * 256 + threadIdx.x;
    if (idx >= NM * NC) return;
    int c = idx % NC;
    int t = (idx / NC) % NT;
    int b = idx / (NC * NT);
    float v;
    if (t < NTT) {
        int tok = tt[b * NTT + t];
        v = tte[(size_t)tok * NC + c] + tpe[t * NC + c];
    } else {
        int ti = t - NTT;
        int tok = it[b * NIT + ti];
        v = ite[(size_t)tok * NC + c] + ipe[ti * NC + c];
    }
    x[idx] = v;
}

// ---------------------------------------------------------------- layernorm
__global__ __launch_bounds__(256) void k_ln(const float* __restrict__ x,
                                            const float* __restrict__ w,
                                            const float* __restrict__ bsh,
                                            unsigned short* __restrict__ out) {
    __shared__ float s1[256], s2[256];
    int tid = threadIdx.x;
    int row = blockIdx.x;
    const float* xr = x + (size_t)row * NC;
    float s = 0.f, ss = 0.f;
    for (int c = tid; c < NC; c += 256) { float v = xr[c]; s += v; ss += v * v; }
    s1[tid] = s; s2[tid] = ss;
    __syncthreads();
    for (int st = 128; st > 0; st >>= 1) {
        if (tid < st) { s1[tid] += s1[tid + st]; s2[tid] += s2[tid + st]; }
        __syncthreads();
    }
    float mean = s1[0] * (1.0f / NC);
    float var = s2[0] * (1.0f / NC) - mean * mean;
    float inv = rsqrtf(var + 1e-5f);
    for (int c = tid; c < NC; c += 256)
        out[(size_t)row * NC + c] = f2bf((xr[c] - mean) * inv * w[c] + bsh[c]);
}

// ------------------------------------------------- weight pack fp32 -> bf16
// Wp layout: [nb = n/128][kb = k/32][8 subtiles][32 lanes][16 slots] ushort,
// i.e. each (nb,kb) B tile is one contiguous 8KB fragment-ordered block.
__global__ __launch_bounds__(256) void k_packw(const float* __restrict__ W,
                                               unsigned short* __restrict__ Wp,
                                               int K, int N) {
    int idx = blockIdx.x * 256 + threadIdx.x;
    if (idx >= K * N) return;
    int kg = idx / N, n = idx % N;
    float v = W[idx];
    int kb = kg >> 5, kk = kg & 31;
    int nb = n >> 7, nl = n & 127;
    size_t base = ((size_t)nb * (K >> 5) + kb) << 12;
    Wp[base + ((nl >> 4) << 9) + (((nl & 15) + (((kk >> 3) & 1) << 4)) << 4) + frag_j(kk)]
        = f2bf(v);
}

// ---------------------------------------------------------------- WMMA GEMM
// C[M,N] = act(A_bf16[M,K] @ Wp + bias) (+resid)   128x128 tile, 8 waves.
// Double-buffered LDS; staging via global_load_async_to_lds_b128 so the
// DMA for tile kb+1 overlaps the 8 WMMAs of tile kb.
// mode 0: fp32 out (+resid); mode 1: bf16 out; mode 2: bf16 scatter [B,H,T,DH]
__global__ __launch_bounds__(256) void k_gemm(const unsigned short* __restrict__ A,
                                              const unsigned short* __restrict__ Wp,
                                              const float* __restrict__ bias,
                                              const float* __restrict__ resid,
                                              float* __restrict__ outF,
                                              unsigned short* __restrict__ outB,
                                              int M, int K, int N, int mode, int act) {
    __shared__ alignas(32) unsigned short sA[2][128 * 32];
    __shared__ alignas(32) unsigned short sB[2][32 * 128];
    int tid = threadIdx.x;
    int lane = tid & 31, wid = tid >> 5;
    int wm = wid & 3, wn = wid >> 2;
    int rowBase = blockIdx.x * 128;
    int colBase = blockIdx.y * 128;
    int nkb = K >> 5;
    size_t wTile0 = ((size_t)(colBase >> 7) * nkb) << 12;  // ushort index of (nb, kb=0)

    v8f acc[2][4];
#pragma unroll
    for (int mt = 0; mt < 2; ++mt)
#pragma unroll
        for (int t = 0; t < 4; ++t) acc[mt][t] = (v8f){0.f,0.f,0.f,0.f,0.f,0.f,0.f,0.f};

    auto issue = [&](int kb, int buf) {
        // A tile: 512 x 16B chunks, swizzled dest; chunk c -> row m=c>>2, group g=c&3
#pragma unroll
        for (int i = 0; i < 2; ++i) {
            int cch = (i << 8) + tid;
            int m = cch >> 2, g = cch & 3;
            const void* src = &A[(size_t)(rowBase + m) * K + (kb << 5) + (g << 3)];
            async_b128(&sA[buf][((m >> 4) << 9)
                                + (((m & 15) + ((g & 1) << 4)) << 4)
                                + ((g >> 1) << 3)], src);
        }
        // B tile: already fragment-ordered & contiguous -> straight 16B copies
#pragma unroll
        for (int i = 0; i < 2; ++i) {
            int cch = (i << 8) + tid;
            async_b128(&sB[buf][cch << 3],
                       &Wp[wTile0 + ((size_t)kb << 12) + ((size_t)cch << 3)]);
        }
    };

    issue(0, 0);
    int buf = 0;
    for (int kb = 0; kb < nkb; ++kb) {
        wait_async0();          // copies for 'buf' landed (issued last iteration)
        __syncthreads();        // and every wave's reads of buf^1 are done
        if (kb + 1 < nkb) issue(kb + 1, buf ^ 1);
        v16bf a0 = *(const v16bf*)&sA[buf][(((wm << 1) + 0) << 9) + (lane << 4)];
        v16bf a1 = *(const v16bf*)&sA[buf][(((wm << 1) + 1) << 9) + (lane << 4)];
#pragma unroll
        for (int t = 0; t < 4; ++t) {
            v16bf bt = *(const v16bf*)&sB[buf][(((wn << 2) + t) << 9) + (lane << 4)];
            acc[0][t] = __builtin_amdgcn_wmma_f32_16x16x32_bf16(false, a0, false, bt,
                                                                (short)0, acc[0][t], false, false);
            acc[1][t] = __builtin_amdgcn_wmma_f32_16x16x32_bf16(false, a1, false, bt,
                                                                (short)0, acc[1][t], false, false);
        }
        buf ^= 1;
    }

    int nloc = lane & 15, half = lane >> 4;
#pragma unroll
    for (int mt = 0; mt < 2; ++mt)
#pragma unroll
        for (int t = 0; t < 4; ++t)
#pragma unroll
            for (int i = 0; i < 8; ++i) {
                int gm = rowBase + wm * 32 + mt * 16 + half * 8 + i;
                int gn = colBase + wn * 64 + t * 16 + nloc;
                float val = acc[mt][t][i];
                if (bias) val += bias[gn];
                if (act) val = 0.5f * val * (1.0f + erff(val * 0.70710678118f));
                if (mode == 0) {
                    size_t o = (size_t)gm * N + gn;
                    if (resid) val += resid[o];
                    outF[o] = val;
                } else if (mode == 1) {
                    outB[(size_t)gm * N + gn] = f2bf(val);
                } else {  // scatter heads: rows (b,t), cols (h,dh) -> [B,H,T,DH]
                    int b = gm / NT, tt = gm % NT;
                    int h = gn >> 6, dh = gn & 63;
                    outB[((size_t)((b * NH + h) * NT + tt) << 6) + dh] = f2bf(val);
                }
            }
}

// ---------------------------------------------------------------- attention
// one wave per (b*H+h, 16-query tile); q,k,v bf16 [B,H,T,DH]; y bf16 [B*T, C]
__global__ __launch_bounds__(32) void k_attn(const unsigned short* __restrict__ q,
                                             const unsigned short* __restrict__ kmat,
                                             const unsigned short* __restrict__ vmat,
                                             unsigned short* __restrict__ y) {
    __shared__ alignas(32) unsigned short ldsP[16 * 32];
    __shared__ alignas(32) unsigned short ldsV[4 * 512];
    int lane = threadIdx.x;
    int bh = blockIdx.x;
    int b = bh / NH, h = bh % NH;
    int q0 = blockIdx.y * 16;
    int nloc = lane & 15, half = lane >> 4;
    const size_t base = (size_t)bh * NT * NDH;

    // Q fragments (scaled by 1/sqrt(DH)): two B128 loads per 32-wide K slice
    Frag aq[2];
#pragma unroll
    for (int ks = 0; ks < 2; ++ks) {
        const unsigned short* qrow = q + base + (size_t)(q0 + nloc) * NDH + ks * 32;
        union { int4 v; unsigned short u[8]; } lo, hi;
        lo.v = *(const int4*)&qrow[half * 8];
        hi.v = *(const int4*)&qrow[16 + half * 8];
#pragma unroll
        for (int j = 0; j < 8; ++j) {
            aq[ks].u[j]     = f2bf(bf2f(lo.u[j]) * 0.125f);
            aq[ks].u[8 + j] = f2bf(bf2f(hi.u[j]) * 0.125f);
        }
    }

    v8f o[4];
#pragma unroll
    for (int t = 0; t < 4; ++t) o[t] = (v8f){0.f,0.f,0.f,0.f,0.f,0.f,0.f,0.f};
    float run_m[8], run_l[8];
#pragma unroll
    for (int i = 0; i < 8; ++i) { run_m[i] = -1e30f; run_l[i] = 0.f; }

    int jv = frag_j(lane);              // V staging: this lane owns key row 'lane'
    int c16 = ((lane >> 3) & 1) << 4;

    int cmax = (q0 + 15) >> 5;
    for (int c = 0; c <= cmax; ++c) {
        int k0 = c << 5;
        // stage V chunk [32 keys x 64 dh] into fragment-swizzled LDS:
        // vectorized global reads (int4 = 8 dh), scattered b16 LDS writes
        {
            const unsigned short* vrow = vmat + base + (size_t)(k0 + lane) * NDH;
#pragma unroll
            for (int g = 0; g < 8; ++g) {
                union { int4 v; unsigned short u[8]; } r;
                r.v = *(const int4*)&vrow[g << 3];
#pragma unroll
                for (int e = 0; e < 8; ++e) {
                    int dh = (g << 3) + e;
                    ldsV[((dh >> 4) << 9) + (((dh & 15) + c16) << 4) + jv] = r.u[e];
                }
            }
        }
        // K fragments: two B128 loads each
        Frag kbf[2][2];
#pragma unroll
        for (int ks = 0; ks < 2; ++ks)
#pragma unroll
            for (int st = 0; st < 2; ++st) {
                const unsigned short* krow =
                    kmat + base + (size_t)(k0 + st * 16 + nloc) * NDH + ks * 32;
                kbf[ks][st].q[0] = *(const int4*)&krow[half * 8];
                kbf[ks][st].q[1] = *(const int4*)&krow[16 + half * 8];
            }
        v8f s0 = (v8f){0.f,0.f,0.f,0.f,0.f,0.f,0.f,0.f};
        v8f s1 = s0;
        s0 = __builtin_amdgcn_wmma_f32_16x16x32_bf16(false, aq[0].v, false, kbf[0][0].v, (short)0, s0, false, false);
        s0 = __builtin_amdgcn_wmma_f32_16x16x32_bf16(false, aq[1].v, false, kbf[1][0].v, (short)0, s0, false, false);
        s1 = __builtin_amdgcn_wmma_f32_16x16x32_bf16(false, aq[0].v, false, kbf[0][1].v, (short)0, s1, false, false);
        s1 = __builtin_amdgcn_wmma_f32_16x16x32_bf16(false, aq[1].v, false, kbf[1][1].v, (short)0, s1, false, false);

        float p0[8], p1[8];
#pragma unroll
        for (int i = 0; i < 8; ++i) {
            int qg = q0 + half * 8 + i;
            float a  = (k0 + nloc      <= qg) ? s0[i] : -1e30f;
            float bb = (k0 + 16 + nloc <= qg) ? s1[i] : -1e30f;
            float mx = fmaxf(a, bb);
            mx = fmaxf(mx, __shfl_xor(mx, 1, 32));
            mx = fmaxf(mx, __shfl_xor(mx, 2, 32));
            mx = fmaxf(mx, __shfl_xor(mx, 4, 32));
            mx = fmaxf(mx, __shfl_xor(mx, 8, 32));
            float nm = fmaxf(run_m[i], mx);
            float fs = __expf(run_m[i] - nm);
            float e0 = __expf(a - nm), e1 = __expf(bb - nm);
            float rs = e0 + e1;
            rs += __shfl_xor(rs, 1, 32);
            rs += __shfl_xor(rs, 2, 32);
            rs += __shfl_xor(rs, 4, 32);
            rs += __shfl_xor(rs, 8, 32);
            run_l[i] = run_l[i] * fs + rs;
            run_m[i] = nm;
            o[0][i] *= fs; o[1][i] *= fs; o[2][i] *= fs; o[3][i] *= fs;
            p0[i] = e0; p1[i] = e1;
        }
        // P -> LDS in A-fragment order
#pragma unroll
        for (int i = 0; i < 8; ++i)
#pragma unroll
            for (int st = 0; st < 2; ++st) {
                int mrow = half * 8 + i;
                int kk = st * 16 + nloc;
                int lp = (mrow & 15) + (((kk >> 3) & 1) << 4);
                ldsP[lp * 16 + frag_j(kk)] = f2bf(st ? p1[i] : p0[i]);
            }
        __syncthreads();
        v16bf pa = *(const v16bf*)&ldsP[lane << 4];
#pragma unroll
        for (int t = 0; t < 4; ++t) {
            v16bf vt = *(const v16bf*)&ldsV[(t << 9) + (lane << 4)];
            o[t] = __builtin_amdgcn_wmma_f32_16x16x32_bf16(false, pa, false, vt, (short)0, o[t], false, false);
        }
        __syncthreads();
    }
#pragma unroll
    for (int t = 0; t < 4; ++t)
#pragma unroll
        for (int i = 0; i < 8; ++i) {
            int mrow = half * 8 + i;
            float ov = o[t][i] / run_l[i];
            y[(size_t)(b * NT + q0 + mrow) * NC + h * NDH + t * 16 + nloc] = f2bf(ov);
        }
}

// ---------------------------------------------------------------- launcher
extern "C" void kernel_launch(void* const* d_in, const int* in_sizes, int n_in,
                              void* d_out, int out_size, void* d_ws, size_t ws_size,
                              hipStream_t stream) {
    const int*   text_tokens = (const int*)d_in[0];
    const int*   image_tokens= (const int*)d_in[1];
    const float* tte   = (const float*)d_in[2];
    const float* ite   = (const float*)d_in[3];
    const float* tpe   = (const float*)d_in[4];
    const float* ipe   = (const float*)d_in[5];
    const float* ln1w  = (const float*)d_in[6];
    const float* ln1b  = (const float*)d_in[7];
    const float* Wq    = (const float*)d_in[8];
    const float* bq    = (const float*)d_in[9];
    const float* Wk    = (const float*)d_in[10];
    const float* bk    = (const float*)d_in[11];
    const float* Wv    = (const float*)d_in[12];
    const float* bv    = (const float*)d_in[13];
    const float* Wo    = (const float*)d_in[14];
    const float* bo    = (const float*)d_in[15];
    const float* ln2w  = (const float*)d_in[16];
    const float* ln2b  = (const float*)d_in[17];
    const float* W1    = (const float*)d_in[18];
    const float* b1    = (const float*)d_in[19];
    const float* W2    = (const float*)d_in[20];
    const float* b2    = (const float*)d_in[21];
    const float* lnfw  = (const float*)d_in[22];
    const float* lnfb  = (const float*)d_in[23];
    const float* headw = (const float*)d_in[24];

    char* ws = (char*)d_ws;
    size_t off = 0;
    float* x = (float*)(ws + off);                     off += (size_t)NM * NC * 4;
    unsigned short* hln = (unsigned short*)(ws + off); off += (size_t)NM * NC * 2;
    unsigned short* qb  = (unsigned short*)(ws + off); off += (size_t)NM * NC * 2;
    unsigned short* kb  = (unsigned short*)(ws + off); off += (size_t)NM * NC * 2;
    unsigned short* vb  = (unsigned short*)(ws + off); off += (size_t)NM * NC * 2;
    unsigned short* yb  = (unsigned short*)(ws + off); off += (size_t)NM * NC * 2;
    unsigned short* h1  = (unsigned short*)(ws + off); off += (size_t)NM * NFF * 2;
    unsigned short* wp  = (unsigned short*)(ws + off); off += (size_t)NC * NV * 2; // max K*N

    k_embed<<<dim3((NM * NC + 255) / 256), 256, 0, stream>>>(
        text_tokens, image_tokens, tte, ite, tpe, ipe, x);

    dim3 g768(NM / 128, NC / 128);
    dim3 gFF(NM / 128, NFF / 128);
    dim3 gHead(NM / 128, NV / 128);
    dim3 gAttn(NB * NH, NT / 16);
    dim3 gPackCC((NC * NC + 255) / 256);
    dim3 gPackFF((NC * NFF + 255) / 256);
    dim3 gPackHd((NC * NV + 255) / 256);

    for (int l = 0; l < NL; ++l) {
        const float* wq = Wq + (size_t)l * NC * NC;
        const float* wk = Wk + (size_t)l * NC * NC;
        const float* wv = Wv + (size_t)l * NC * NC;
        const float* wo = Wo + (size_t)l * NC * NC;
        const float* w1 = W1 + (size_t)l * NC * NFF;
        const float* w2 = W2 + (size_t)l * NFF * NC;

        k_ln<<<NM, 256, 0, stream>>>(x, ln1w + l * NC, ln1b + l * NC, hln);
        k_packw<<<gPackCC, 256, 0, stream>>>(wq, wp, NC, NC);
        k_gemm<<<g768, 256, 0, stream>>>(hln, wp, bq + l * NC, nullptr, nullptr, qb,
                                         NM, NC, NC, 2, 0);
        k_packw<<<gPackCC, 256, 0, stream>>>(wk, wp, NC, NC);
        k_gemm<<<g768, 256, 0, stream>>>(hln, wp, bk + l * NC, nullptr, nullptr, kb,
                                         NM, NC, NC, 2, 0);
        k_packw<<<gPackCC, 256, 0, stream>>>(wv, wp, NC, NC);
        k_gemm<<<g768, 256, 0, stream>>>(hln, wp, bv + l * NC, nullptr, nullptr, vb,
                                         NM, NC, NC, 2, 0);
        k_attn<<<gAttn, 32, 0, stream>>>(qb, kb, vb, yb);
        k_packw<<<gPackCC, 256, 0, stream>>>(wo, wp, NC, NC);
        k_gemm<<<g768, 256, 0, stream>>>(yb, wp, bo + l * NC, x, x, nullptr,
                                         NM, NC, NC, 0, 0);
        k_ln<<<NM, 256, 0, stream>>>(x, ln2w + l * NC, ln2b + l * NC, hln);
        k_packw<<<gPackFF, 256, 0, stream>>>(w1, wp, NC, NFF);
        k_gemm<<<gFF, 256, 0, stream>>>(hln, wp, b1 + l * NFF, nullptr, nullptr, h1,
                                        NM, NC, NFF, 1, 1);
        k_packw<<<gPackFF, 256, 0, stream>>>(w2, wp, NFF, NC);
        k_gemm<<<g768, 256, 0, stream>>>(h1, wp, b2 + l * NC, x, x, nullptr,
                                         NM, NFF, NC, 0, 0);
    }
    k_ln<<<NM, 256, 0, stream>>>(x, lnfw, lnfb, hln);
    k_packw<<<gPackHd, 256, 0, stream>>>(headw, wp, NC, NV);
    k_gemm<<<gHead, 256, 0, stream>>>(hln, wp, nullptr, nullptr, (float*)d_out, nullptr,
                                      NM, NC, NV, 0, 0);
}
